// SparseWindowAttention_27273042330372
// MI455X (gfx1250) — matrix operand order repaired and checked by reading, not compiled, not run on hardware
//
#include <hip/hip_runtime.h>

// ---------------------------------------------------------------------------
// MI455X (gfx1250) implementation of SparseWindowAttention.
// All heavy matmuls use v_wmma_f32_16x16x32_bf16 (wave32 WMMA).
// Softmax cross-lane reductions use DPP16 ROW_XMASK (pure VALU, no LDS).
// ---------------------------------------------------------------------------

typedef __bf16 bf16;
typedef __attribute__((ext_vector_type(16))) __bf16 v16bf;
typedef __attribute__((ext_vector_type(8)))  __bf16 v8bf;
typedef __attribute__((ext_vector_type(8)))  float   v8f;

static __device__ __forceinline__ v16bf cat16(v8bf a, v8bf b) {
  return __builtin_shufflevector(a, b, 0,1,2,3,4,5,6,7,8,9,10,11,12,13,14,15);
}
static __device__ __forceinline__ v8f wmma_bf16(v16bf a, v16bf b, v8f c) {
  return __builtin_amdgcn_wmma_f32_16x16x32_bf16(false, a, false, b, (short)0, c, false, false);
}

// DPP16 ROW_XMASK lane permute: lane <- lane ^ M within each row of 16.
// Pure-VALU cross-lane reduction (no ds_bpermute / dscnt waits), and it
// stays inside the 16-lane half-groups that match WMMA fragment rows.
template <int M>
static __device__ __forceinline__ float dpp_xor(float x) {
  int v = __builtin_amdgcn_update_dpp(0, __float_as_int(x),
                                      0x160 | M, 0xf, 0xf, true);
  return __int_as_float(v);
}
static __device__ __forceinline__ float redmax16(float v) {
  v = fmaxf(v, dpp_xor<1>(v));
  v = fmaxf(v, dpp_xor<2>(v));
  v = fmaxf(v, dpp_xor<4>(v));
  v = fmaxf(v, dpp_xor<8>(v));
  return v;
}
static __device__ __forceinline__ float redsum16(float v) {
  v += dpp_xor<1>(v);
  v += dpp_xor<2>(v);
  v += dpp_xor<4>(v);
  v += dpp_xor<8>(v);
  return v;
}

// Geometry constants
#define TT    4
#define HH    60
#define WW    72
#define CC    512
#define NHEAD 4
#define CH    128
#define NWIN  96
#define WTOK  45          // 5*9 window tokens
#define NPOOL 270         // 15*18 pooled tokens
#define NKW   463         // 45 + 148 + 270 keys per (window, t)
#define NKEYS 1852        // TT * NKW
#define NKB   58          // ceil(1852/32)
#define NTOK  17280       // TT*HH*WW
#define HWTOK 4320        // HH*WW
#define SCALE 0.08838834764831845f  // 1/sqrt(128)

// --------------------------- conversion kernels ----------------------------

__global__ void convert_x_k(const float* __restrict__ x, bf16* __restrict__ xb, int n) {
  int g = blockIdx.x * blockDim.x + threadIdx.x;
  if (g < n) xb[g] = (bf16)x[g];
}

// WT[which][n][k] = W_which[k][n] in bf16 (transposed so WMMA B-fragment
// loads 16 contiguous halves of one output-column per lane).
__global__ void convert_wT_k(const float* __restrict__ Wq, const float* __restrict__ Wk,
                             const float* __restrict__ Wv, const float* __restrict__ Wo,
                             bf16* __restrict__ WT) {
  int g = blockIdx.x * blockDim.x + threadIdx.x;   // 4*512*512 threads
  int which = g >> 18;
  int rem = g & 262143;
  int nn = rem >> 9, k = rem & 511;
  const float* W = (which == 0) ? Wq : (which == 1) ? Wk : (which == 2) ? Wv : Wo;
  WT[(size_t)which * 262144 + (size_t)nn * 512 + k] = (bf16)W[(size_t)k * 512 + nn];
}

// Depthwise 4x4 pooling -> bf16 (input to pooled K/V GEMM)
__global__ void pool_k(const float* __restrict__ x, const float* __restrict__ pw,
                       const float* __restrict__ pbias, bf16* __restrict__ pxb) {
  int g = blockIdx.x * blockDim.x + threadIdx.x;   // 1080*512
  if (g >= 1080 * 512) return;
  int row = g >> 9, c = g & 511;
  int t = row / 270, rp = row % 270, py = rp / 18, px = rp % 18;
  float s = pbias[c];
  #pragma unroll
  for (int i = 0; i < 4; ++i)
    #pragma unroll
    for (int j = 0; j < 4; ++j)
      s += x[((size_t)((t * HH + py * 4 + i) * WW + px * 4 + j)) * CC + c] * pw[c * 16 + i * 4 + j];
  pxb[g] = (bf16)s;
}

// Per-window gather table (463 key sources) + mask flags.
// Entry v < 4320 : spatial token (h*72+w);  v >= 4320 : pooled token v-4320.
__global__ void win_index_k(const float* __restrict__ mask, int* __restrict__ widx,
                            int* __restrict__ wflag) {
  int w = threadIdx.x;
  if (w >= NWIN) return;
  int wy = w >> 3, wx = w & 7;
  int base = w * NKW, slot = 0;
  for (int i = 0; i < 5; ++i)
    for (int j = 0; j < 9; ++j)
      widx[base + slot++] = (wy * 5 + i) * WW + (wx * 9 + j);
  const int sh[4] = {-3, -3, 3, 3};
  const int sw[4] = {-5,  5, -5, 5};
  for (int b = 0; b < 4; ++b)
    for (int i = 0; i < 5; ++i)
      for (int j = 0; j < 9; ++j) {
        bool inval = (b == 0 && i < 2 && j < 4) || (b == 1 && i < 2 && j >= 5) ||
                     (b == 2 && i >= 3 && j < 4) || (b == 3 && i >= 3 && j >= 5);
        if (inval) continue;
        int srcH = (wy * 5 + i - sh[b] + 2 * HH) % HH;   // roll(k, s): src = pos - s
        int srcW = (wx * 9 + j - sw[b] + 2 * WW) % WW;
        widx[base + slot++] = srcH * WW + srcW;
      }
  for (int p = 0; p < NPOOL; ++p) widx[base + slot++] = HWTOK + p;
  float s = 0.f;
  for (int t = 0; t < TT; ++t) {
    float mx = 0.f;
    for (int i = 0; i < 5; ++i)
      for (int j = 0; j < 9; ++j)
        mx = fmaxf(mx, mask[(size_t)((t * HH + wy * 5 + i) * WW + wx * 9 + j)]);
    s += mx;
  }
  wflag[w] = (s > 0.f) ? 1 : 0;
}

// ------------------------------ WMMA GEMM ----------------------------------
// Y[M][512] = A[M][512] (bf16) @ W (bf16, stored transposed BT[512][512]) + bias
// Block: 256 threads / 8 waves, 128x128 output tile, B-panel staged in LDS.
template <bool OUTF32>
__global__ __launch_bounds__(256) void gemm512_k(const bf16* __restrict__ A,
                                                 const bf16* __restrict__ BT,
                                                 const float* __restrict__ bias,
                                                 void* __restrict__ outp, int M) {
  __shared__ bf16 Bp[128][32];   // 8KB panel: 128 out-cols x 32 k-halves
  int tid = threadIdx.x;
  int wave = tid >> 5, lane = tid & 31, n = lane & 15, hi = lane >> 4;
  int row0 = blockIdx.x * 128;
  int col0 = blockIdx.y * 128;
  int arow = row0 + wave * 16 + n;
  if (arow >= M) arow = M - 1;          // clamp; stores are guarded
  const bf16* aptr = A + (size_t)arow * CC;
  v8f acc[8];
  #pragma unroll
  for (int ct = 0; ct < 8; ++ct)
    #pragma unroll
    for (int r = 0; r < 8; ++r) acc[ct][r] = 0.f;

  for (int k0 = 0; k0 < 512; k0 += 32) {
    // stage 128x32 B panel (512 8-half segs, 2 per thread)
    int s0 = tid * 2;
    {
      int rB = s0 >> 2, cp = (s0 & 3) * 8;
      *(v8bf*)&Bp[rB][cp] = *(const v8bf*)(BT + (size_t)(col0 + rB) * 512 + k0 + cp);
      if (k0 < 480)   // prefetch next panel segment into L0/L2
        __builtin_prefetch(BT + (size_t)(col0 + rB) * 512 + k0 + 32 + cp, 0, 1);
      int s1 = s0 + 1;
      rB = s1 >> 2; cp = (s1 & 3) * 8;
      *(v8bf*)&Bp[rB][cp] = *(const v8bf*)(BT + (size_t)(col0 + rB) * 512 + k0 + cp);
    }
    __syncthreads();
    v16bf af = cat16(*(const v8bf*)(aptr + k0 + hi * 16),
                     *(const v8bf*)(aptr + k0 + hi * 16 + 8));
    #pragma unroll
    for (int ct = 0; ct < 8; ++ct) {
      v16bf bf_ = cat16(*(const v8bf*)&Bp[ct * 16 + n][hi * 16],
                        *(const v8bf*)&Bp[ct * 16 + n][hi * 16 + 8]);
      acc[ct] = wmma_bf16(af, bf_, acc[ct]);
    }
    __syncthreads();
  }

  #pragma unroll
  for (int ct = 0; ct < 8; ++ct) {
    int col = col0 + ct * 16 + n;
    float b = bias[col];
    #pragma unroll
    for (int r = 0; r < 8; ++r) {
      int row = row0 + wave * 16 + hi * 8 + r;
      if (row < M) {
        float val = acc[ct][r] + b;
        if (OUTF32) ((float*)outp)[(size_t)row * CC + col] = val;
        else        ((bf16*)outp)[(size_t)row * CC + col] = (bf16)val;
      }
    }
  }
}

// ------------------------ mixed-window flash attention ---------------------
// One block per (window, head): 12 waves, one 16-query tile each (180->192).
// Online softmax over 58 key-blocks of 32; K and V^T staged in LDS.
__global__ __launch_bounds__(384) void attn_mixed_k(
    const bf16* __restrict__ qb, const bf16* __restrict__ kb, const bf16* __restrict__ vb,
    const bf16* __restrict__ pkb, const bf16* __restrict__ pvb,
    const int* __restrict__ widx, const int* __restrict__ wflag, bf16* __restrict__ yb) {
  int w = blockIdx.x, head = blockIdx.y;
  if (!wflag[w]) return;                 // standard path handles this window
  int wy = w >> 3, wx = w & 7;

  __shared__ bf16 Ks[32][128];           // 32 keys x 128 ch
  __shared__ bf16 Vt[128][32];           // transposed V tile: ch x key
  __shared__ bf16 Pb[12][16][32];        // per-wave P re-layout buffer

  int tid = threadIdx.x;
  int wave = tid >> 5, lane = tid & 31, n = lane & 15, hi = lane >> 4;

  // A-fragment query row = wave*16 + n (clamped into valid range)
  int qa = wave * 16 + n; if (qa > 179) qa = 179;
  int tq = qa / WTOK, uq = qa % WTOK;
  const bf16* qptr = qb + ((size_t)(tq * HWTOK + (wy * 5 + uq / 9) * WW + (wx * 9 + uq % 9))) * CC
                        + head * CH;

  float mrow[8], lrow[8], alpha[8];
  v8f o[8];
  #pragma unroll
  for (int r = 0; r < 8; ++r) { mrow[r] = -1e30f; lrow[r] = 0.f; }
  #pragma unroll
  for (int ct = 0; ct < 8; ++ct)
    #pragma unroll
    for (int r = 0; r < 8; ++r) o[ct][r] = 0.f;

  for (int kb0 = 0; kb0 < NKB; ++kb0) {
    int k0 = kb0 * 32;
    // cooperative gather of 32 keys into LDS (K row-major, V transposed)
    for (int sidx = tid; sidx < 512; sidx += 384) {
      int kr = sidx >> 4, cp = (sidx & 15) * 8;
      int kk = k0 + kr; if (kk > NKEYS - 1) kk = NKEYS - 1;   // pad: masked later
      int tt = kk / NKW, jj = kk % NKW;
      int v = widx[w * NKW + jj];
      const bf16 *ksrc, *vsrc;
      if (v < HWTOK) {
        size_t off = ((size_t)(tt * HWTOK + v)) * CC + head * CH + cp;
        ksrc = kb + off; vsrc = vb + off;
      } else {
        size_t off = ((size_t)(tt * NPOOL + (v - HWTOK))) * CC + head * CH + cp;
        ksrc = pkb + off; vsrc = pvb + off;
      }
      *(v8bf*)&Ks[kr][cp] = *(const v8bf*)ksrc;
      v8bf dv = *(const v8bf*)vsrc;
      #pragma unroll
      for (int e = 0; e < 8; ++e) Vt[cp + e][kr] = dv[e];
    }
    __syncthreads();

    // S = Q @ K^T for two 16x16 key tiles
    v8f s1, s2;
    #pragma unroll
    for (int r = 0; r < 8; ++r) { s1[r] = 0.f; s2[r] = 0.f; }
    #pragma unroll
    for (int c = 0; c < 4; ++c) {
      v16bf aq = cat16(*(const v8bf*)(qptr + c * 32 + hi * 16),
                       *(const v8bf*)(qptr + c * 32 + hi * 16 + 8));
      v16bf bk1 = cat16(*(const v8bf*)&Ks[n][c * 32 + hi * 16],
                        *(const v8bf*)&Ks[n][c * 32 + hi * 16 + 8]);
      v16bf bk2 = cat16(*(const v8bf*)&Ks[16 + n][c * 32 + hi * 16],
                        *(const v8bf*)&Ks[16 + n][c * 32 + hi * 16 + 8]);
      s1 = wmma_bf16(aq, bk1, s1);
      s2 = wmma_bf16(aq, bk2, s2);
    }

    // online softmax update; row stats reduced over 16-lane half-groups
    // with DPP ROW_XMASK (VALU-only, matches WMMA fragment row mapping)
    int key1 = k0 + n, key2 = k0 + 16 + n;
    #pragma unroll
    for (int r = 0; r < 8; ++r) {
      float f1 = (key1 < NKEYS) ? s1[r] * SCALE : -1e30f;
      float f2 = (key2 < NKEYS) ? s2[r] * SCALE : -1e30f;
      float vmax = redmax16(fmaxf(f1, f2));
      float mnew = fmaxf(mrow[r], vmax);
      float a2 = __expf(mrow[r] - mnew);
      float p1 = __expf(f1 - mnew), p2 = __expf(f2 - mnew);
      float ps = redsum16(p1 + p2);
      lrow[r] = lrow[r] * a2 + ps;
      mrow[r] = mnew;
      alpha[r] = a2;
      Pb[wave][hi * 8 + r][n]      = (bf16)p1;
      Pb[wave][hi * 8 + r][16 + n] = (bf16)p2;
    }
    #pragma unroll
    for (int ct = 0; ct < 8; ++ct)
      #pragma unroll
      for (int r = 0; r < 8; ++r) o[ct][r] *= alpha[r];

    __builtin_amdgcn_wave_barrier();   // keep P store->load ordered (same wave)

    // O += P @ V  (A = P fragment re-read from LDS, B = V^T tiles)
    v16bf ap = cat16(*(const v8bf*)&Pb[wave][n][hi * 16],
                     *(const v8bf*)&Pb[wave][n][hi * 16 + 8]);
    #pragma unroll
    for (int ct = 0; ct < 8; ++ct) {
      v16bf bv2 = cat16(*(const v8bf*)&Vt[ct * 16 + n][hi * 16],
                        *(const v8bf*)&Vt[ct * 16 + n][hi * 16 + 8]);
      o[ct] = wmma_bf16(ap, bv2, o[ct]);
    }
    __syncthreads();
  }

  // write normalized output (scatter into (t,h,w,c) bf16 buffer)
  #pragma unroll
  for (int r = 0; r < 8; ++r) {
    int qd = wave * 16 + hi * 8 + r;
    if (qd >= 180) continue;
    int td = qd / WTOK, ud = qd % WTOK;
    bf16* dst = yb + ((size_t)(td * HWTOK + (wy * 5 + ud / 9) * WW + (wx * 9 + ud % 9))) * CC
                   + head * CH;
    float inv = 1.f / lrow[r];
    #pragma unroll
    for (int ct = 0; ct < 8; ++ct) dst[ct * 16 + n] = (bf16)(o[ct][r] * inv);
  }
}

// ---------------------- standard (per-t) window attention ------------------
// <2% of FLOPs: scalar VALU path, LDS-staged Q/K/V, 45x45 per (w,head,t).
__global__ __launch_bounds__(64) void attn_std_k(
    const bf16* __restrict__ qb, const bf16* __restrict__ kb, const bf16* __restrict__ vb,
    const int* __restrict__ wflag, bf16* __restrict__ yb) {
  int w = blockIdx.x, head = blockIdx.y, t = blockIdx.z;
  if (wflag[w]) return;                  // mixed path handles this window
  int wy = w >> 3, wx = w & 7;
  __shared__ bf16 Qs[WTOK][CH], Ksh[WTOK][CH], Vs[WTOK][CH];
  int tid = threadIdx.x;
  for (int s = tid; s < WTOK * 16; s += 64) {
    int r = s >> 4, cp = (s & 15) * 8;
    int gh = wy * 5 + r / 9, gw = wx * 9 + r % 9;
    size_t off = ((size_t)(t * HWTOK + gh * WW + gw)) * CC + head * CH + cp;
    *(v8bf*)&Qs[r][cp]  = *(const v8bf*)(qb + off);
    *(v8bf*)&Ksh[r][cp] = *(const v8bf*)(kb + off);
    *(v8bf*)&Vs[r][cp]  = *(const v8bf*)(vb + off);
  }
  __syncthreads();
  int q = tid;
  if (q >= WTOK) return;
  float s[WTOK];
  float mx = -1e30f;
  for (int k = 0; k < WTOK; ++k) {
    float acc = 0.f;
    for (int c = 0; c < CH; ++c) acc += (float)Qs[q][c] * (float)Ksh[k][c];
    acc *= SCALE;
    s[k] = acc; mx = fmaxf(mx, acc);
  }
  float l = 0.f;
  for (int k = 0; k < WTOK; ++k) { s[k] = __expf(s[k] - mx); l += s[k]; }
  float inv = 1.f / l;
  int gh = wy * 5 + q / 9, gw = wx * 9 + q % 9;
  bf16* dst = yb + ((size_t)(t * HWTOK + gh * WW + gw)) * CC + head * CH;
  for (int c = 0; c < CH; ++c) {
    float acc = 0.f;
    for (int k = 0; k < WTOK; ++k) acc += s[k] * (float)Vs[k][c];
    dst[c] = (bf16)(acc * inv);
  }
}

// ------------------------------- launcher ----------------------------------

extern "C" void kernel_launch(void* const* d_in, const int* in_sizes, int n_in,
                              void* d_out, int out_size, void* d_ws, size_t ws_size,
                              hipStream_t stream) {
  (void)in_sizes; (void)n_in; (void)out_size; (void)ws_size;
  const float* x     = (const float*)d_in[0];
  const float* mask  = (const float*)d_in[1];
  const float* Wq    = (const float*)d_in[2];
  const float* bq    = (const float*)d_in[3];
  const float* Wk    = (const float*)d_in[4];
  const float* bk    = (const float*)d_in[5];
  const float* Wv    = (const float*)d_in[6];
  const float* bv    = (const float*)d_in[7];
  const float* Wo    = (const float*)d_in[8];
  const float* bo    = (const float*)d_in[9];
  const float* pw    = (const float*)d_in[10];
  const float* pbias = (const float*)d_in[11];

  char* base = (char*)d_ws;
  size_t off = 0;
  auto alloc = [&](size_t bytes) -> char* {
    char* r = base + off;
    off += bytes;
    off = (off + 255) & ~((size_t)255);
    return r;
  };
  bf16* xb   = (bf16*)alloc((size_t)NTOK * CC * 2);
  bf16* WT   = (bf16*)alloc((size_t)4 * 512 * 512 * 2);
  bf16* qb2  = (bf16*)alloc((size_t)NTOK * CC * 2);
  bf16* kb2  = (bf16*)alloc((size_t)NTOK * CC * 2);
  bf16* vb2  = (bf16*)alloc((size_t)NTOK * CC * 2);
  bf16* pxb  = (bf16*)alloc((size_t)1080 * CC * 2);
  bf16* pkb  = (bf16*)alloc((size_t)1080 * CC * 2);
  bf16* pvb  = (bf16*)alloc((size_t)1080 * CC * 2);
  bf16* yb   = (bf16*)alloc((size_t)NTOK * CC * 2);
  int*  widx = (int*)alloc((size_t)NWIN * NKW * 4);
  int*  wflag= (int*)alloc((size_t)NWIN * 4);
  bf16* WqT = WT;
  bf16* WkT = WT + 262144;
  bf16* WvT = WT + 2 * 262144;
  bf16* WoT = WT + 3 * 262144;

  convert_x_k<<<34560, 256, 0, stream>>>(x, xb, NTOK * CC);
  convert_wT_k<<<4096, 256, 0, stream>>>(Wq, Wk, Wv, Wo, WT);
  pool_k<<<2160, 256, 0, stream>>>(x, pw, pbias, pxb);
  win_index_k<<<1, 96, 0, stream>>>(mask, widx, wflag);

  gemm512_k<false><<<dim3(135, 4), 256, 0, stream>>>(xb, WqT, bq, qb2, NTOK);
  gemm512_k<false><<<dim3(135, 4), 256, 0, stream>>>(xb, WkT, bk, kb2, NTOK);
  gemm512_k<false><<<dim3(135, 4), 256, 0, stream>>>(xb, WvT, bv, vb2, NTOK);
  gemm512_k<false><<<dim3(9, 4),   256, 0, stream>>>(pxb, WkT, bk, pkb, 1080);
  gemm512_k<false><<<dim3(9, 4),   256, 0, stream>>>(pxb, WvT, bv, pvb, 1080);

  attn_mixed_k<<<dim3(NWIN, NHEAD), 384, 0, stream>>>(qb2, kb2, vb2, pkb, pvb, widx, wflag, yb);
  attn_std_k<<<dim3(NWIN, NHEAD, TT), 64, 0, stream>>>(qb2, kb2, vb2, wflag, yb);

  gemm512_k<true><<<dim3(135, 4), 256, 0, stream>>>(yb, WoT, bo, d_out, NTOK);
}
